// SpGraphAttentionLayer_17695265259688
// MI455X (gfx1250) — matrix operand order repaired and checked by reading, not compiled, not run on hardware
//
#include <hip/hip_runtime.h>
#include <hip/hip_bf16.h>
#include <stdint.h>

#define LRELU_SLOPE 0.2f

typedef __attribute__((ext_vector_type(2)))  float    f32x2;
typedef __attribute__((ext_vector_type(4)))  float    f32x4;
typedef __attribute__((ext_vector_type(8)))  float    f32x8;
typedef __attribute__((ext_vector_type(2)))  int      i32x2;
typedef __attribute__((ext_vector_type(4)))  int      i32x4;
typedef __attribute__((ext_vector_type(2)))  unsigned u32x2;
typedef __attribute__((ext_vector_type(4)))  unsigned u32x4;
typedef __attribute__((ext_vector_type(8)))  unsigned u32x8;
typedef __attribute__((ext_vector_type(16))) __bf16   bf16x16;

__device__ __forceinline__ unsigned f32_to_bf16_bits(float f) {
  unsigned u = __builtin_bit_cast(unsigned, f);
  u += 0x7FFFu + ((u >> 16) & 1u);   // round-to-nearest-even
  return u >> 16;
}

// --------------------------- async LDS staging probe -----------------------
#if defined(__has_builtin)
#if __has_builtin(__builtin_amdgcn_global_load_async_to_lds_b128)
#define HAVE_ASYNC_LDS 1
#endif
#endif

#ifdef HAVE_ASYNC_LDS
typedef __attribute__((address_space(1))) i32x4 as1_v4i;   // global int4
typedef __attribute__((address_space(3))) i32x4 as3_v4i;   // LDS int4
__device__ __forceinline__ void async_ld16(const void* g, void* l) {
  // 16B per lane, Memory -> LDS, tracked by ASYNCcnt
  __builtin_amdgcn_global_load_async_to_lds_b128(
      (as1_v4i*)(uintptr_t)g, (as3_v4i*)(unsigned)(uintptr_t)l, 0, 0);
}
#if __has_builtin(__builtin_amdgcn_s_wait_asynccnt)
#define WAIT_ASYNC(n) __builtin_amdgcn_s_wait_asynccnt(n)
#else
#define WAIT_ASYNC(n) asm volatile("s_wait_asynccnt %0" ::"n"(n) : "memory")
#endif
#endif

// ---------------------------------------------------------------------------
// Kernel A: Hh = h (16384x256) * W (256x256), f32 WMMA 16x16x4.
// Each block: 16 rows x 128 cols (8 waves, one 16x16 tile each).
// Writes Hh (f32 row-major) and HhT (bf16, [B][F_OUT][N] column-major).
// ---------------------------------------------------------------------------
__global__ __launch_bounds__(256) void k_gemm_hW(
    const float* __restrict__ h, const float* __restrict__ W,
    float* __restrict__ Hh, unsigned short* __restrict__ HhT) {
  const int lane = threadIdx.x & 31;
  const int wave = threadIdx.x >> 5;
  const int hi   = lane >> 4;
  const int l16  = lane & 15;
  const int rowBase = blockIdx.x * 16;
  const int col     = blockIdx.y * 128 + wave * 16 + l16;

  const float* __restrict__ hrow = h + (size_t)(rowBase + l16) * 256;
  f32x8 acc = {};
  for (int k = 0; k < 256; k += 4) {
    f32x2 A = *(const f32x2*)(hrow + k + 2 * hi);       // K pair per half-wave
    f32x2 B;
    B.x = W[(size_t)(k + 2 * hi) * 256 + col];
    B.y = W[(size_t)(k + 2 * hi + 1) * 256 + col];
    acc = __builtin_amdgcn_wmma_f32_16x16x4_f32(false, A, false, B,
                                                (short)0, acc, false, false);
  }

  const int b  = rowBase >> 12;
  const int n0 = (rowBase & 4095) + hi * 8;
  u32x4 pk;
  for (int j = 0; j < 4; ++j)
    pk[j] = f32_to_bf16_bits(acc[2 * j]) | (f32_to_bf16_bits(acc[2 * j + 1]) << 16);
  *(u32x4*)(HhT + ((size_t)(b * 256 + col) * 4096 + n0)) = pk;
  for (int v = 0; v < 8; ++v)
    Hh[(size_t)(rowBase + hi * 8 + v) * 256 + col] = acc[v];
}

// ---------------------------------------------------------------------------
// Kernel B: s = Hh . a_src, d = Hh . a_dst (per row). One row per wave.
// ---------------------------------------------------------------------------
__global__ __launch_bounds__(256) void k_attn_sd(
    const float* __restrict__ Hh, const float* __restrict__ a,
    float* __restrict__ s, float* __restrict__ d) {
  const int lane = threadIdx.x & 31;
  const int wave = threadIdx.x >> 5;
  const int row  = blockIdx.x * 8 + wave;
  const float* __restrict__ hr = Hh + (size_t)row * 256;
  float ss = 0.0f, dd = 0.0f;
  for (int j = 0; j < 8; ++j) {
    int c = lane + 32 * j;
    float v = hr[c];
    ss += v * a[c];
    dd += v * a[256 + c];
  }
  for (int off = 16; off > 0; off >>= 1) {
    ss += __shfl_xor(ss, off, 32);
    dd += __shfl_xor(dd, off, 32);
  }
  if (lane == 0) { s[row] = ss; d[row] = dd; }
}

// ---------------------------------------------------------------------------
// Kernel C: out = elu( (e @ Hh) / rowsum(e) ).
// One block per (batch, 32-row strip) covering all 256 columns, so adj
// streams exactly once (NT loads + prefetch 1 chunk ahead).
// Per 32-m chunk:
//   - 256 threads build the 32x32 bf16 e-tile in LDS (4 exps each),
//   - the 16KB HhT chunk is async-staged to LDS (double buffered,
//     GLOBAL_LOAD_ASYNC_TO_LDS_B128 + s_wait_asynccnt) when available,
//   - 8 waves each run 4x v_wmma_f32_16x16x32_bf16 (2 row-groups x 2 col
//     tiles, B operands reused across row groups).
// ---------------------------------------------------------------------------
__global__ __launch_bounds__(256) void k_gat_aggregate(
    const int* __restrict__ adj, const unsigned short* __restrict__ HhT,
    const float* __restrict__ s, const float* __restrict__ d,
    float* __restrict__ out) {
  __shared__ __align__(16) unsigned e_lds[32 * 16];     // 32 rows x 32 m, bf16
  __shared__ __align__(16) float rowsum_lds[32];
#ifdef HAVE_ASYNC_LDS
  __shared__ __align__(16) unsigned short bstage[2][256 * 32];  // 2 x 16 KB
#endif

  const int tid  = threadIdx.x;
  const int lane = tid & 31;
  const int wave = tid >> 5;
  const int hi   = lane >> 4;
  const int l16  = lane & 15;
  const int b       = blockIdx.x >> 7;          // 128 strips per batch
  const int rowBase = (blockIdx.x & 127) * 32;

  // producer mapping: thread owns (erow, em..em+3) of the 32x32 e tile
  const int erow = tid >> 3;              // 0..31
  const int em   = (tid & 7) * 4;         // 0,4,..,28
  const float srow = s[b * 4096 + rowBase + erow];
  const int*   __restrict__ adjRow = adj + (size_t)(b * 4096 + rowBase + erow) * 4096;
  const float* __restrict__ dRow   = d + b * 4096;

  // consumer mapping: wave owns columns [c0, c0+32)
  const int c0 = wave * 32;
  const unsigned short* __restrict__ hbase = HhT + (size_t)(b * 256) * 4096;

  if (tid < 32) rowsum_lds[tid] = 0.0f;
  float racc = 0.0f;
  f32x8 acc00 = {}, acc01 = {}, acc10 = {}, acc11 = {};

#ifdef HAVE_ASYNC_LDS
  {  // stage chunk 0: thread t stages column t (64B = 4 x async b128)
    const unsigned short* g = hbase + (size_t)tid * 4096;
    unsigned short* l = &bstage[0][tid * 32];
    for (int j = 0; j < 4; ++j) async_ld16(g + j * 8, l + j * 8);
  }
#endif
  __syncthreads();

  for (int it = 0; it < 128; ++it) {
    const int m0 = it * 32;

    // ---- produce 4 e values (adj streamed once: NT load + prefetch) ----
    i32x4 av = __builtin_nontemporal_load((const i32x4*)(adjRow + m0 + em));
    f32x4 dv = *(const f32x4*)(dRow + m0 + em);
    if (it + 1 < 128) __builtin_prefetch(adjRow + m0 + 32 + em, 0, 0);
    float e0, e1, e2, e3;
    {
      float l0 = srow + dv.x, l1 = srow + dv.y, l2 = srow + dv.z, l3 = srow + dv.w;
      l0 = (l0 > 0.0f) ? l0 : LRELU_SLOPE * l0;
      l1 = (l1 > 0.0f) ? l1 : LRELU_SLOPE * l1;
      l2 = (l2 > 0.0f) ? l2 : LRELU_SLOPE * l2;
      l3 = (l3 > 0.0f) ? l3 : LRELU_SLOPE * l3;
      e0 = av.x ? __expf(-l0) : 0.0f;
      e1 = av.y ? __expf(-l1) : 0.0f;
      e2 = av.z ? __expf(-l2) : 0.0f;
      e3 = av.w ? __expf(-l3) : 0.0f;
    }
    racc += (e0 + e1) + (e2 + e3);
    u32x2 epk = { f32_to_bf16_bits(e0) | (f32_to_bf16_bits(e1) << 16),
                  f32_to_bf16_bits(e2) | (f32_to_bf16_bits(e3) << 16) };
    ((u32x2*)e_lds)[tid] = epk;      // row erow, m em..em+3

#ifdef HAVE_ASYNC_LDS
    if (it + 1 < 128) {   // stage next chunk into the other buffer
      const unsigned short* g = hbase + (size_t)tid * 4096 + (m0 + 32);
      unsigned short* l = &bstage[(it + 1) & 1][tid * 32];
      for (int j = 0; j < 4; ++j) async_ld16(g + j * 8, l + j * 8);
      WAIT_ASYNC(4);      // retire chunk `it`, keep chunk `it+1` in flight
    } else {
      WAIT_ASYNC(0);
    }
#endif
    __syncthreads();

    // ---- A operands from e tile (contiguous k = 16*hi + s, same order as B) ----
    const u32x4* eb = (const u32x4*)e_lds;       // row stride = 4 x 16B
    u32x4 a0lo = eb[l16 * 4 + 2 * hi];
    u32x4 a0hi = eb[l16 * 4 + 2 * hi + 1];
    u32x4 a1lo = eb[(16 + l16) * 4 + 2 * hi];
    u32x4 a1hi = eb[(16 + l16) * 4 + 2 * hi + 1];
    u32x8 a0p = { a0lo.x, a0lo.y, a0lo.z, a0lo.w, a0hi.x, a0hi.y, a0hi.z, a0hi.w };
    u32x8 a1p = { a1lo.x, a1lo.y, a1lo.z, a1lo.w, a1hi.x, a1hi.y, a1hi.z, a1hi.w };
    bf16x16 A0 = __builtin_bit_cast(bf16x16, a0p);
    bf16x16 A1 = __builtin_bit_cast(bf16x16, a1p);

    // ---- B operands (shared by both row groups) ----
#ifdef HAVE_ASYNC_LDS
    const unsigned short* p0 = &bstage[it & 1][(c0 + l16) * 32 + hi * 16];
    const unsigned short* p1 = &bstage[it & 1][(c0 + 16 + l16) * 32 + hi * 16];
#else
    const unsigned short* p0 = hbase + (size_t)(c0 + l16) * 4096 + m0 + hi * 16;
    const unsigned short* p1 = hbase + (size_t)(c0 + 16 + l16) * 4096 + m0 + hi * 16;
#endif
    u32x4 b0a = *(const u32x4*)p0;
    u32x4 b0b = *(const u32x4*)(p0 + 8);
    u32x4 b1a = *(const u32x4*)p1;
    u32x4 b1b = *(const u32x4*)(p1 + 8);
    u32x8 b0p = { b0a.x, b0a.y, b0a.z, b0a.w, b0b.x, b0b.y, b0b.z, b0b.w };
    u32x8 b1p = { b1a.x, b1a.y, b1a.z, b1a.w, b1b.x, b1b.y, b1b.z, b1b.w };
    bf16x16 B0 = __builtin_bit_cast(bf16x16, b0p);
    bf16x16 B1 = __builtin_bit_cast(bf16x16, b1p);

    acc00 = __builtin_amdgcn_wmma_f32_16x16x32_bf16(false, A0, false, B0, (short)0, acc00, false, false);
    acc01 = __builtin_amdgcn_wmma_f32_16x16x32_bf16(false, A0, false, B1, (short)0, acc01, false, false);
    acc10 = __builtin_amdgcn_wmma_f32_16x16x32_bf16(false, A1, false, B0, (short)0, acc10, false, false);
    acc11 = __builtin_amdgcn_wmma_f32_16x16x32_bf16(false, A1, false, B1, (short)0, acc11, false, false);
    __syncthreads();     // protect e_lds / bstage before next chunk
  }

  // ---- rowsum reduction (8 threads per row contributed partials) ----
  atomicAdd(&rowsum_lds[erow], racc);
  __syncthreads();

  const f32x4* rv = (const f32x4*)rowsum_lds;
  f32x4 q0 = rv[hi * 2 + 0], q1 = rv[hi * 2 + 1];         // rows 0..15 group
  f32x4 q2 = rv[4 + hi * 2 + 0], q3 = rv[4 + hi * 2 + 1]; // rows 16..31 group
  float rs0[8] = { q0.x, q0.y, q0.z, q0.w, q1.x, q1.y, q1.z, q1.w };
  float rs1[8] = { q2.x, q2.y, q2.z, q2.w, q3.x, q3.y, q3.z, q3.w };

  for (int v = 0; v < 8; ++v) {
    float r0 = rs0[v]; r0 = (r0 == 0.0f) ? 1.0f : r0;
    float r1 = rs1[v]; r1 = (r1 == 0.0f) ? 1.0f : r1;
    size_t row0 = (size_t)(b * 4096 + rowBase + hi * 8 + v) * 256;
    size_t row1 = (size_t)(b * 4096 + rowBase + 16 + hi * 8 + v) * 256;
    float x;
    x = acc00[v] / r0; out[row0 + c0 + l16]      = (x > 0.0f) ? x : (__expf(x) - 1.0f);
    x = acc01[v] / r0; out[row0 + c0 + 16 + l16] = (x > 0.0f) ? x : (__expf(x) - 1.0f);
    x = acc10[v] / r1; out[row1 + c0 + l16]      = (x > 0.0f) ? x : (__expf(x) - 1.0f);
    x = acc11[v] / r1; out[row1 + c0 + 16 + l16] = (x > 0.0f) ? x : (__expf(x) - 1.0f);
  }
}

// ---------------------------------------------------------------------------
extern "C" void kernel_launch(void* const* d_in, const int* in_sizes, int n_in,
                              void* d_out, int out_size, void* d_ws, size_t ws_size,
                              hipStream_t stream) {
  const float* h   = (const float*)d_in[0];   // (4,4096,256) f32
  const int*   adj = (const int*)  d_in[1];   // (4,4096,4096) i32
  const float* W   = (const float*)d_in[2];   // (256,256) f32
  const float* a   = (const float*)d_in[3];   // (1,512) f32
  float* out = (float*)d_out;                 // (4,4096,256) f32

  // workspace: Hh f32 (16 MB) | HhT bf16 (8 MB) | s (64 KB) | d (64 KB)
  char* ws = (char*)d_ws;
  float*          Hh  = (float*)ws;
  unsigned short* HhT = (unsigned short*)(ws + (size_t)16384 * 256 * 4);
  float*          s   = (float*)(ws + (size_t)16384 * 256 * 6);
  float*          d   = s + 16384;

  dim3 gA(1024, 2);
  k_gemm_hW<<<gA, 256, 0, stream>>>(h, W, Hh, HhT);
  k_attn_sd<<<2048, 256, 0, stream>>>(Hh, a, s, d);
  k_gat_aggregate<<<512, 256, 0, stream>>>(adj, HhT, s, d, out);
}
